// LSTM_16269336117662
// MI455X (gfx1250) — compile-verified, matrix operand order checked by reading, not run on hardware
//
#include <hip/hip_runtime.h>

// LSTM for MI455X (gfx1250, wave32, WMMA).
// Phase 0: convert x / weights f32 -> bf16 (storage only; accumulate in f32).
// Phase 1: x@Wx^T + bx for all 4 gates via v_wmma_f32_16x16x32_bf16, output
//          stored bf16 pre-swizzled into WMMA C-fragment order.
// Phase 2: persistent scan kernel: 8 workgroups (one per 16 batch rows, the
//          WMMA M granularity) run all 1024 steps; h shared via LDS (bf16),
//          c and gate accumulators stay in VGPRs; workgroup barriers only.

typedef __attribute__((ext_vector_type(16))) __bf16          v16bf;
typedef __attribute__((ext_vector_type(8)))  float           v8f;
typedef __attribute__((ext_vector_type(8)))  unsigned short  v8us;

#define S_LEN 1024
#define B_SZ  128
#define I_DIM 256
#define H_DIM 256

__device__ __forceinline__ unsigned short f32_to_bf16(float f) {
  union { float f; unsigned int u; } v; v.f = f;
  unsigned int r = v.u + 0x7FFFu + ((v.u >> 16) & 1u);   // round to nearest even
  return (unsigned short)(r >> 16);
}
__device__ __forceinline__ float bf16_to_f32(unsigned short h) {
  union { unsigned int u; float f; } v; v.u = ((unsigned int)h) << 16;
  return v.f;
}
__device__ __forceinline__ float sigmoid_f(float x) {
  return 1.0f / (1.0f + __expf(-x));
}
__device__ __forceinline__ float tanh_f(float x) {
  float e = __expf(-2.0f * fabsf(x));
  float t = (1.0f - e) / (1.0f + e);
  return copysignf(t, x);
}

__global__ void cvt_f32_bf16(const float* __restrict__ src,
                             unsigned short* __restrict__ dst, int n) {
  int i = blockIdx.x * blockDim.x + threadIdx.x;
  if (i < n) dst[i] = f32_to_bf16(src[i]);
}

// ---------------------------------------------------------------------------
// Phase 1: gate pre-activations  xg = x @ Wx[g]^T + bx[g]   (bf16 out)
// One wave per 16x16 output tile; K-loop of 8 WMMAs (K=256).
// Output swizzled: xg[((s*8+btile)*4+g)*16+ntile][lane][r]  (v8us per lane).
// ---------------------------------------------------------------------------
__global__ __launch_bounds__(256)
void lstm_xproj(const unsigned short* __restrict__ xb,    // [S*B, I] bf16
                const unsigned short* __restrict__ wxb,   // 4 x [H, I] bf16
                const float* __restrict__ bx0, const float* __restrict__ bx1,
                const float* __restrict__ bx2, const float* __restrict__ bx3,
                unsigned short* __restrict__ xg)
{
  int lane  = threadIdx.x & 31;
  int wid   = blockIdx.x * 8 + (threadIdx.x >> 5);  // 524288 waves total
  int g     = wid & 3;
  int t     = wid >> 2;
  int ntile = t & 15;
  int mtile = t >> 4;                               // 0..8191 (rows of S*B / 16)
  int half  = lane >> 4;
  int l16   = lane & 15;

  // A: rows of x (contiguous K, lane-half selects K sub-chunk).
  const unsigned short* arow =
      xb + (size_t)(mtile * 16 + l16) * I_DIM + half * 16;
  // B: column n of Wx^T == row n of Wx (contiguous K), same K permutation.
  const unsigned short* brow =
      wxb + (size_t)g * (H_DIM * I_DIM) + (size_t)(ntile * 16 + l16) * I_DIM + half * 16;

  v8f acc = {};
  #pragma unroll
  for (int kk = 0; kk < 8; ++kk) {
    v16bf a = *reinterpret_cast<const v16bf*>(arow + kk * 32);
    v16bf b = *reinterpret_cast<const v16bf*>(brow + kk * 32);
    acc = __builtin_amdgcn_wmma_f32_16x16x32_bf16(false, a, false, b,
                                                  (short)0, acc, false, false);
  }

  const float* bx = (g == 0) ? bx0 : (g == 1) ? bx1 : (g == 2) ? bx2 : bx3;
  float bv = bx[ntile * 16 + l16];

  int s     = mtile >> 3;   // 128/16 = 8 mtiles per timestep
  int btile = mtile & 7;

  v8us o;
  #pragma unroll
  for (int r = 0; r < 8; ++r) o[r] = f32_to_bf16(acc[r] + bv);
  size_t idx = ((((size_t)(s * 8 + btile) * 4 + g) * 16 + ntile) * 32 + lane) * 8;
  *reinterpret_cast<v8us*>(xg + idx) = o;   // one coalesced 16B store per lane
}

// ---------------------------------------------------------------------------
// Phase 2: the scan. 8 workgroups x 512 threads (16 waves).
// Workgroup = 16 batch rows; wave = 16 hidden columns, all 4 gates.
// ---------------------------------------------------------------------------
__global__ __launch_bounds__(512)
void lstm_scan(const unsigned short* __restrict__ xg,
               const unsigned short* __restrict__ whb,   // 4 x [H,H] bf16
               const float* __restrict__ bh0, const float* __restrict__ bh1,
               const float* __restrict__ bh2, const float* __restrict__ bh3,
               float* __restrict__ out)
{
  __shared__ __align__(32) unsigned short h_lds[16][H_DIM];  // 8 KB bf16 hidden

  int tid   = threadIdx.x;
  int lane  = tid & 31;
  int wave  = tid >> 5;            // 0..15 -> ntile (output column block)
  int btile = blockIdx.x;          // 0..7  -> batch row block
  int half  = lane >> 4;
  int l16   = lane & 15;
  int col   = wave * 16 + l16;     // hidden column owned by this lane

  for (int i = tid; i < 16 * H_DIM; i += blockDim.x)
    (&h_lds[0][0])[i] = 0;         // h0 = 0
  __syncthreads();

  float bh[4] = { bh0[col], bh1[col], bh2[col], bh3[col] };

  const unsigned short* wh[4];
  #pragma unroll
  for (int g = 0; g < 4; ++g)
    wh[g] = whb + (size_t)g * (H_DIM * H_DIM) + (size_t)col * H_DIM + half * 16;

  v8f c_acc = {};
  v8f hval  = {};
  int rowbase = btile * 16 + 8 * half;   // global batch row = rowbase + r

  for (int s = 0; s < S_LEN; ++s) {
    // Init accumulators from precomputed x-projections (coalesced 16B loads),
    // prefetch next step's fragments into cache.
    v8f acc[4];
    int sn = (s + 1 < S_LEN) ? s + 1 : s;
    #pragma unroll
    for (int g = 0; g < 4; ++g) {
      size_t idx  = ((((size_t)(s  * 8 + btile) * 4 + g) * 16 + wave) * 32 + lane) * 8;
      size_t idxn = ((((size_t)(sn * 8 + btile) * 4 + g) * 16 + wave) * 32 + lane) * 8;
      __builtin_prefetch(xg + idxn, 0, 1);           // global_prefetch_b8
      v8us v = *reinterpret_cast<const v8us*>(xg + idx);
      #pragma unroll
      for (int r = 0; r < 8; ++r) acc[g][r] = bf16_to_f32(v[r]);
    }

    // h_prev @ Wh[g]^T : A from LDS, B from L2-resident weights.
    #pragma unroll
    for (int kk = 0; kk < 8; ++kk) {
      v16bf a = *reinterpret_cast<const v16bf*>(&h_lds[l16][kk * 32 + half * 16]);
      #pragma unroll
      for (int g = 0; g < 4; ++g) {
        v16bf b = *reinterpret_cast<const v16bf*>(wh[g] + kk * 32);
        acc[g] = __builtin_amdgcn_wmma_f32_16x16x32_bf16(false, a, false, b,
                                                         (short)0, acc[g],
                                                         false, false);
      }
    }

    // Gate activations + cell update, all in registers (same (b,h) ownership
    // every step, so c never leaves VGPRs).
    #pragma unroll
    for (int r = 0; r < 8; ++r) {
      float gi = sigmoid_f(acc[0][r] + bh[0]);
      float gf = sigmoid_f(acc[1][r] + bh[1]);
      float go = sigmoid_f(acc[2][r] + bh[2]);
      float gc = tanh_f   (acc[3][r] + bh[3]);
      float c  = gf * c_acc[r] + gi * gc;
      c_acc[r] = c;
      hval[r]  = go * tanh_f(c);
    }

    __syncthreads();   // everyone done reading h_lds for this step
    #pragma unroll
    for (int r = 0; r < 8; ++r) {
      h_lds[r + 8 * half][col] = f32_to_bf16(hval[r]);
      out[((size_t)s * B_SZ + rowbase + r) * H_DIM + col] = hval[r];
    }
    __syncthreads();   // h_lds ready for next step's A loads
  }

  // Final (h, c) appended after h_seq in d_out.
  float* hfin = out + (size_t)S_LEN * B_SZ * H_DIM;
  float* cfin = hfin + (size_t)B_SZ * H_DIM;
  #pragma unroll
  for (int r = 0; r < 8; ++r) {
    hfin[(size_t)(rowbase + r) * H_DIM + col] = hval[r];
    cfin[(size_t)(rowbase + r) * H_DIM + col] = c_acc[r];
  }
}

extern "C" void kernel_launch(void* const* d_in, const int* in_sizes, int n_in,
                              void* d_out, int out_size, void* d_ws, size_t ws_size,
                              hipStream_t stream)
{
  (void)in_sizes; (void)n_in; (void)out_size; (void)ws_size;
  // setup_inputs order: x, (Wxi,Whi,bxi,bhi), (Wxf,Whf,bxf,bhf),
  //                        (Wxo,Who,bxo,bho), (Wxc,Whc,bxc,bhc)
  const float* x     = (const float*)d_in[0];
  const float* Wx[4] = {(const float*)d_in[1], (const float*)d_in[5],
                        (const float*)d_in[9], (const float*)d_in[13]};
  const float* Wh[4] = {(const float*)d_in[2], (const float*)d_in[6],
                        (const float*)d_in[10], (const float*)d_in[14]};
  const float* bx[4] = {(const float*)d_in[3], (const float*)d_in[7],
                        (const float*)d_in[11], (const float*)d_in[15]};
  const float* bh[4] = {(const float*)d_in[4], (const float*)d_in[8],
                        (const float*)d_in[12], (const float*)d_in[16]};
  float* out = (float*)d_out;

  // Workspace layout (~321 MB total):
  unsigned short* xb  = (unsigned short*)d_ws;                 // x bf16: 64 MiB
  unsigned short* wxb = xb  + (size_t)S_LEN * B_SZ * I_DIM;    // 4x[H,I] bf16
  unsigned short* whb = wxb + (size_t)4 * H_DIM * I_DIM;       // 4x[H,H] bf16
  unsigned short* xg  = whb + (size_t)4 * H_DIM * H_DIM;       // 256 MiB bf16

  int nx = S_LEN * B_SZ * I_DIM;
  cvt_f32_bf16<<<(nx + 255) / 256, 256, 0, stream>>>(x, xb, nx);
  for (int g = 0; g < 4; ++g) {
    cvt_f32_bf16<<<(H_DIM * I_DIM + 255) / 256, 256, 0, stream>>>(
        Wx[g], wxb + (size_t)g * H_DIM * I_DIM, H_DIM * I_DIM);
    cvt_f32_bf16<<<(H_DIM * H_DIM + 255) / 256, 256, 0, stream>>>(
        Wh[g], whb + (size_t)g * H_DIM * H_DIM, H_DIM * H_DIM);
  }

  // 8192 mtiles * 16 ntiles * 4 gates = 524288 waves, 8 waves/block.
  lstm_xproj<<<65536, 256, 0, stream>>>(xb, wxb, bx[0], bx[1], bx[2], bx[3], xg);

  // Whole recurrence in one kernel: 8 workgroups, 16 waves each.
  lstm_scan<<<8, 512, 0, stream>>>(xg, whb, bh[0], bh[1], bh[2], bh[3], out);
}
